// VitBlock_48730698940993
// MI455X (gfx1250) — matrix operand order repaired and checked by reading, not compiled, not run on hardware
//
#include <hip/hip_runtime.h>
#include <hip/hip_bf16.h>

// ---------------------------------------------------------------------------
// ViT block forward for MI455X (gfx1250, wave32, WMMA).
// All GEMMs + both attention einsums run on v_wmma_f32_16x16x32_f16.
// Round 2: 2x M register blocking in GEMMs (B-fragment reuse, dual WMMA
// chains per load) + global_prefetch on streaming A rows.
// ---------------------------------------------------------------------------

typedef __attribute__((ext_vector_type(16))) _Float16 v16h;
typedef __attribute__((ext_vector_type(8)))  float    v8f;

union FragU { v16h v; uint4 q[2]; };

// A-fragment (16x32 f16, lane = row M, per ISA 7.12.2):
//   element e<8 : K = g*8 + e ; element e>=8 : K = 16 + g*8 + (e-8)
__device__ __forceinline__ v16h load_afrag(const _Float16* p, int g) {
    FragU f;
    f.q[0] = *(const uint4*)(p + g * 8);
    f.q[1] = *(const uint4*)(p + 16 + g * 8);
    return f.v;
}
// B-fragment (32x16 f16, lane = col N, same K order as A-fragment:
// e<8 : K = g*8 + e ; e>=8 : K = 16 + g*8 + (e-8)). p points at Bt row (i.e. B column).
__device__ __forceinline__ v16h load_bfrag(const _Float16* p, int g) {
    FragU f;
    f.q[0] = *(const uint4*)(p + g * 8);
    f.q[1] = *(const uint4*)(p + 16 + g * 8);
    return f.v;
}
__device__ __forceinline__ v8f wmma32(v16h a, v16h b, v8f c) {
    return __builtin_amdgcn_wmma_f32_16x16x32_f16(
        false, a, false, b, (short)0, c, false, false);
}
__device__ __forceinline__ float gelu_f(float x) {
    return 0.5f * x * (1.0f + erff(x * 0.70710678118654752f));
}

// ---------------------------- constants ------------------------------------
#define TOK   8192   // 4 * 2048 tokens
#define SEQ   2048
#define CDIM  768
#define NH    12
#define HD    64
#define HID   3072
#define NQKV  2304
#define PFH   256    // prefetch distance in halves (8 k-iterations ahead)

// ------------------------ weight convert + transpose -----------------------
// W [K x N] f32 row-major  ->  Wt [N x K] f16 row-major
__global__ void convT_kernel(const float* __restrict__ W, _Float16* __restrict__ Wt,
                             int K, int N) {
    size_t total = (size_t)K * N;
    for (size_t i = blockIdx.x * 256ull + threadIdx.x; i < total;
         i += (size_t)gridDim.x * 256ull) {
        int n = (int)(i % N);
        int kk = (int)(i / N);
        Wt[(size_t)n * K + kk] = (_Float16)W[i];
    }
}

// ------------------------------ LayerNorm ----------------------------------
// one 256-thread block per token row of 768; fp32 stats, f16 output
__global__ void ln_kernel(const float* __restrict__ x, const float* __restrict__ gam,
                          const float* __restrict__ bet, _Float16* __restrict__ out) {
    int row = blockIdx.x;
    const float* xr = x + (size_t)row * CDIM;
    int c0 = threadIdx.x, c1 = c0 + 256, c2 = c0 + 512;
    float v0 = xr[c0], v1 = xr[c1], v2 = xr[c2];
    float s = v0 + v1 + v2;
    float s2 = v0 * v0 + v1 * v1 + v2 * v2;
    for (int off = 16; off; off >>= 1) {
        s  += __shfl_xor(s,  off, 32);
        s2 += __shfl_xor(s2, off, 32);
    }
    __shared__ float red[2][8];
    int wave = threadIdx.x >> 5, lane = threadIdx.x & 31;
    if (lane == 0) { red[0][wave] = s; red[1][wave] = s2; }
    __syncthreads();
    float ts = 0.f, ts2 = 0.f;
#pragma unroll
    for (int i = 0; i < 8; ++i) { ts += red[0][i]; ts2 += red[1][i]; }
    float mean = ts * (1.0f / CDIM);
    float var  = ts2 * (1.0f / CDIM) - mean * mean;
    float rstd = rsqrtf(var + 1e-5f);
    _Float16* orow = out + (size_t)row * CDIM;
    orow[c0] = (_Float16)((v0 - mean) * rstd * gam[c0] + bet[c0]);
    orow[c1] = (_Float16)((v1 - mean) * rstd * gam[c1] + bet[c1]);
    orow[c2] = (_Float16)((v2 - mean) * rstd * gam[c2] + bet[c2]);
}

// ---------------------- generic WMMA GEMM (wave = 32x64) -------------------
// C[M,N] = A[M,K](f16) @ Bt[N,K]^T(f16) + bias (+resid) (+gelu)
// Each wave: 2 M-tiles x 4 N-tiles -> every B-fragment feeds 2 WMMAs.
template<bool GELU, bool RES, bool OUTH>
__global__ void gemm_kernel(const _Float16* __restrict__ A, const _Float16* __restrict__ Bt,
                            const float* __restrict__ bias, const float* __restrict__ resid,
                            _Float16* __restrict__ outH, float* __restrict__ outF,
                            int N, int K) {
    int lane = threadIdx.x & 31, wave = threadIdx.x >> 5;
    int n16 = lane & 15, g = lane >> 4;
    int mBase = blockIdx.y * 256 + wave * 32;
    int nBase = blockIdx.x * 64;
    const _Float16* aRow0 = A + (size_t)(mBase + n16) * K;
    const _Float16* aRow1 = A + (size_t)(mBase + 16 + n16) * K;
    v8f acc[2][4] = {};
    for (int kk = 0; kk < K; kk += 32) {
        __builtin_prefetch(aRow0 + kk + PFH, 0, 1);
        __builtin_prefetch(aRow1 + kk + PFH, 0, 1);
        v16h af0 = load_afrag(aRow0 + kk, g);
        v16h af1 = load_afrag(aRow1 + kk, g);
#pragma unroll
        for (int t = 0; t < 4; ++t) {
            const _Float16* bRow = Bt + (size_t)(nBase + t * 16 + n16) * K + kk;
            v16h bf = load_bfrag(bRow, g);
            acc[0][t] = wmma32(af0, bf, acc[0][t]);
            acc[1][t] = wmma32(af1, bf, acc[1][t]);
        }
    }
#pragma unroll
    for (int u = 0; u < 2; ++u)
#pragma unroll
        for (int t = 0; t < 4; ++t) {
            int col = nBase + t * 16 + n16;
            float bv = bias[col];
#pragma unroll
            for (int r = 0; r < 8; ++r) {
                int row = mBase + u * 16 + r + 8 * g;
                float v = acc[u][t][r] + bv;
                if (RES)  v += resid[(size_t)row * N + col];
                if (GELU) v = gelu_f(v);
                if (OUTH) outH[(size_t)row * N + col] = (_Float16)v;
                else      outF[(size_t)row * N + col] = v;
            }
        }
}

// -------------------- QKV GEMM with scatter epilogue -----------------------
// q -> [b,h,seq,64] (pre-scaled by head_dim^-0.5), k -> [b,h,seq,64],
// v -> transposed [b,h,64,seq] so attention PV B-fragments are contiguous.
__global__ void gemm_qkv_kernel(const _Float16* __restrict__ A, const _Float16* __restrict__ Bt,
                                const float* __restrict__ bias,
                                _Float16* __restrict__ qo, _Float16* __restrict__ ko,
                                _Float16* __restrict__ vto) {
    const int K = CDIM;
    int lane = threadIdx.x & 31, wave = threadIdx.x >> 5;
    int n16 = lane & 15, g = lane >> 4;
    int mBase = blockIdx.y * 256 + wave * 32;
    int nBase = blockIdx.x * 64;
    const _Float16* aRow0 = A + (size_t)(mBase + n16) * K;
    const _Float16* aRow1 = A + (size_t)(mBase + 16 + n16) * K;
    v8f acc[2][4] = {};
    for (int kk = 0; kk < K; kk += 32) {
        __builtin_prefetch(aRow0 + kk + PFH, 0, 1);
        __builtin_prefetch(aRow1 + kk + PFH, 0, 1);
        v16h af0 = load_afrag(aRow0 + kk, g);
        v16h af1 = load_afrag(aRow1 + kk, g);
#pragma unroll
        for (int t = 0; t < 4; ++t) {
            const _Float16* bRow = Bt + (size_t)(nBase + t * 16 + n16) * K + kk;
            v16h bf = load_bfrag(bRow, g);
            acc[0][t] = wmma32(af0, bf, acc[0][t]);
            acc[1][t] = wmma32(af1, bf, acc[1][t]);
        }
    }
#pragma unroll
    for (int u = 0; u < 2; ++u)
#pragma unroll
        for (int t = 0; t < 4; ++t) {
            int col = nBase + t * 16 + n16;
            float bv = bias[col];
#pragma unroll
            for (int r = 0; r < 8; ++r) {
                int rowTok = mBase + u * 16 + r + 8 * g;
                int bb = rowTok >> 11, tok = rowTok & (SEQ - 1);
                float v = acc[u][t][r] + bv;
                if (col < CDIM) {
                    int hh = col >> 6, d = col & 63;
                    qo[(((size_t)bb * NH + hh) * SEQ + tok) * HD + d] = (_Float16)(v * 0.125f);
                } else if (col < 2 * CDIM) {
                    int c = col - CDIM;
                    int hh = c >> 6, d = c & 63;
                    ko[(((size_t)bb * NH + hh) * SEQ + tok) * HD + d] = (_Float16)v;
                } else {
                    int c = col - 2 * CDIM;
                    int hh = c >> 6, d = c & 63;
                    vto[(((size_t)bb * NH + hh) * HD + d) * SEQ + tok] = (_Float16)v;
                }
            }
        }
}

// --------------------------- flash attention -------------------------------
// grid (SEQ/128, B*NH); each wave: 16 q rows, online softmax over key tiles
// of 32, P staged via LDS to re-layout D-tile -> A-fragment.
__global__ void attn_kernel(const _Float16* __restrict__ q, const _Float16* __restrict__ k,
                            const _Float16* __restrict__ vt, _Float16* __restrict__ ctx) {
    int bh = blockIdx.y;
    int b = bh / NH, h = bh % NH;
    int lane = threadIdx.x & 31, wave = threadIdx.x >> 5;
    int n16 = lane & 15, g = lane >> 4;
    int qBase = blockIdx.x * 128 + wave * 16;

    const _Float16* qRow = q + ((size_t)bh * SEQ + qBase + n16) * HD;
    v16h qa0 = load_afrag(qRow, g);        // q dims 0..31
    v16h qa1 = load_afrag(qRow + 32, g);   // q dims 32..63

    v8f o[4] = {};
    float rm[8], rl[8];
#pragma unroll
    for (int r = 0; r < 8; ++r) { rm[r] = -1e30f; rl[r] = 0.f; }

    __shared__ __align__(16) _Float16 pbuf[8][16][32];

    for (int kb = 0; kb < SEQ; kb += 32) {
        v8f s0 = {}, s1 = {};
        {   // keys kb .. kb+15
            const _Float16* kr = k + ((size_t)bh * SEQ + kb + n16) * HD;
            s0 = wmma32(qa0, load_bfrag(kr, g), s0);
            s0 = wmma32(qa1, load_bfrag(kr + 32, g), s0);
        }
        {   // keys kb+16 .. kb+31
            const _Float16* kr = k + ((size_t)bh * SEQ + kb + 16 + n16) * HD;
            s1 = wmma32(qa0, load_bfrag(kr, g), s1);
            s1 = wmma32(qa1, load_bfrag(kr + 32, g), s1);
        }
        float alpha[8];
#pragma unroll
        for (int r = 0; r < 8; ++r) {
            float a0 = s0[r], a1 = s1[r];
            float cm = fmaxf(a0, a1);
            for (int off = 1; off < 16; off <<= 1)
                cm = fmaxf(cm, __shfl_xor(cm, off, 32));
            float mn = fmaxf(rm[r], cm);
            alpha[r] = __expf(rm[r] - mn);
            rm[r] = mn;
            float p0 = __expf(a0 - mn), p1 = __expf(a1 - mn);
            float ps = p0 + p1;
            for (int off = 1; off < 16; off <<= 1)
                ps += __shfl_xor(ps, off, 32);
            rl[r] = rl[r] * alpha[r] + ps;
            pbuf[wave][r + 8 * g][n16]      = (_Float16)p0;
            pbuf[wave][r + 8 * g][n16 + 16] = (_Float16)p1;
        }
        asm volatile("s_wait_dscnt 0" ::: "memory");
        v16h pf = load_afrag(&pbuf[wave][n16][0], g);   // P as 16x32 A-fragment
#pragma unroll
        for (int t = 0; t < 4; ++t) {
#pragma unroll
            for (int r = 0; r < 8; ++r) o[t][r] *= alpha[r];
            const _Float16* vr = vt + ((size_t)bh * HD + t * 16 + n16) * SEQ + kb;
            o[t] = wmma32(pf, load_bfrag(vr, g), o[t]);
        }
    }
#pragma unroll
    for (int t = 0; t < 4; ++t)
#pragma unroll
        for (int r = 0; r < 8; ++r) {
            int row = qBase + r + 8 * g;
            ctx[((size_t)b * SEQ + row) * CDIM + h * HD + t * 16 + n16] =
                (_Float16)(o[t][r] / rl[r]);
        }
}

// ------------------------------ launcher -----------------------------------
extern "C" void kernel_launch(void* const* d_in, const int* in_sizes, int n_in,
                              void* d_out, int out_size, void* d_ws, size_t ws_size,
                              hipStream_t stream) {
    const float* x      = (const float*)d_in[0];
    const float* qkv_w  = (const float*)d_in[1];
    const float* qkv_b  = (const float*)d_in[2];
    const float* proj_w = (const float*)d_in[3];
    const float* proj_b = (const float*)d_in[4];
    const float* fc1_w  = (const float*)d_in[5];
    const float* fc1_b  = (const float*)d_in[6];
    const float* fc2_w  = (const float*)d_in[7];
    const float* fc2_b  = (const float*)d_in[8];
    const float* ln1_g  = (const float*)d_in[9];
    const float* ln1_b  = (const float*)d_in[10];
    const float* ln2_g  = (const float*)d_in[11];
    const float* ln2_b  = (const float*)d_in[12];
    float* out = (float*)d_out;

    // workspace carve-up (all offsets 4 KB-aligned)
    char* ws = (char*)d_ws;
    _Float16* h1    = (_Float16*)(ws);                      // 8192x768    f16
    _Float16* qkvWt = (_Float16*)(ws + 12582912);           // 2304x768    f16
    _Float16* projWt= (_Float16*)(ws + 16121856);           // 768x768     f16
    _Float16* fc1Wt = (_Float16*)(ws + 17301504);           // 3072x768    f16
    _Float16* fc2Wt = (_Float16*)(ws + 22020096);           // 768x3072    f16
    _Float16* qbuf  = (_Float16*)(ws + 26738688);           // b,h,seq,64  f16
    _Float16* kbuf  = (_Float16*)(ws + 39321600);           // b,h,seq,64  f16
    _Float16* vtbuf = (_Float16*)(ws + 51904512);           // b,h,64,seq  f16
    _Float16* ctx   = (_Float16*)(ws + 64487424);           // 8192x768    f16
    float*    x1    = (float*)   (ws + 77070336);           // 8192x768    f32
    _Float16* h2    = (_Float16*)(ws + 102236160);          // 8192x768    f16
    _Float16* act   = (_Float16*)(ws + 114819072);          // 8192x3072   f16

    dim3 blk(256);

    // weight transposes (f32 -> f16 [NxK])
    convT_kernel<<<2048, blk, 0, stream>>>(qkv_w, qkvWt, CDIM, NQKV);
    convT_kernel<<<1024, blk, 0, stream>>>(proj_w, projWt, CDIM, CDIM);
    convT_kernel<<<2048, blk, 0, stream>>>(fc1_w, fc1Wt, CDIM, HID);
    convT_kernel<<<2048, blk, 0, stream>>>(fc2_w, fc2Wt, HID, CDIM);

    // LN1
    ln_kernel<<<TOK, blk, 0, stream>>>(x, ln1_g, ln1_b, h1);

    // QKV projection + scatter (q scaled, v transposed)
    gemm_qkv_kernel<<<dim3(NQKV / 64, TOK / 256), blk, 0, stream>>>(
        h1, qkvWt, qkv_b, qbuf, kbuf, vtbuf);

    // attention
    attn_kernel<<<dim3(SEQ / 128, 4 * NH), blk, 0, stream>>>(qbuf, kbuf, vtbuf, ctx);

    // proj + residual(x) -> x1 (fp32)
    gemm_kernel<false, true, false><<<dim3(CDIM / 64, TOK / 256), blk, 0, stream>>>(
        ctx, projWt, proj_b, x, nullptr, x1, CDIM, CDIM);

    // LN2
    ln_kernel<<<TOK, blk, 0, stream>>>(x1, ln2_g, ln2_b, h2);

    // FC1 + exact GELU -> f16
    gemm_kernel<true, false, true><<<dim3(HID / 64, TOK / 256), blk, 0, stream>>>(
        h2, fc1Wt, fc1_b, nullptr, act, nullptr, HID, CDIM);

    // FC2 + residual(x1) -> out (fp32)
    gemm_kernel<false, true, false><<<dim3(CDIM / 64, TOK / 256), blk, 0, stream>>>(
        act, fc2Wt, fc2_b, x1, nullptr, out, CDIM, HID);
}
